// MultiscaleVectorQuantizer_51453708206307
// MI455X (gfx1250) — matrix operand order repaired and checked by reading, not compiled, not run on hardware
//
#include <hip/hip_runtime.h>

typedef __attribute__((ext_vector_type(16))) _Float16 v16h;
typedef __attribute__((ext_vector_type(8)))  _Float16 v8h;
typedef __attribute__((ext_vector_type(8)))  float    v8f;

#define B_   32
#define C_   64
#define H_   16
#define W_   16
#define K_   8192
#define IMG  (B_*C_*H_*W_)     // 524288 elements
#define CONV_K 576             // 64 * 3 * 3
#define NPOS (B_*H_*W_)        // 8192 conv output rows

#define VQ_TC 128              // codewords staged per LDS buffer (16 KB)
#define VQ_STAGES (K_/VQ_TC)   // 64

// workspace layout (byte offsets into d_ws)
#define OFF_ZREST 0u                            // 2 MB  float [B][C][16][16]
#define OFF_EMB16 (2u<<20)                      // 1 MB  f16   [K][64]
#define OFF_WSQ   (3u<<20)                      // 32 KB float [K]
#define OFF_TOK   ((3u<<20)+(64u<<10))          // 32 KB int   [K]
#define OFF_WF16  ((3u<<20)+(128u<<10))         // 72 KB f16   [64][576]
#define OFF_ZF16  (4u<<20)                      // 1 MB  f16   [8192][64]
#define OFF_HUP   (5u<<20)                      // 2 MB  float [B][C][16][16]
#define OFF_COL   (7u<<20)                      // 9 MB  f16   [8192][576]

#if __has_builtin(__builtin_amdgcn_global_load_async_to_lds_b128) && \
    __has_builtin(__builtin_amdgcn_s_wait_asynccnt)
#define HAVE_ASYNC_LDS 1
#else
#define HAVE_ASYNC_LDS 0
#endif

static __device__ __forceinline__ v16h cat8(v8h lo, v8h hi) {
  return __builtin_shufflevector(lo, hi, 0,1,2,3,4,5,6,7,8,9,10,11,12,13,14,15);
}

#if HAVE_ASYNC_LDS
typedef int v4i_vs __attribute__((vector_size(16)));
static __device__ __forceinline__ void async_b128(const void* g, void* l) {
  // builtin prototype (from compiler diagnostic): (int4*, int4*, imm offset, imm cpol)
  __builtin_amdgcn_global_load_async_to_lds_b128((v4i_vs*)g, (v4i_vs*)l, 0, 0);
}
#endif

// copy 16 KB (codebook stage) global -> LDS with a full 256-thread block
static __device__ __forceinline__ void stage_cb(_Float16* dst, const _Float16* src) {
#if HAVE_ASYNC_LDS
  const char* g = (const char*)src + threadIdx.x * 16;
  char*       l = (char*)dst + threadIdx.x * 16;
#pragma unroll
  for (int j = 0; j < 4; ++j) async_b128(g + j * 4096, l + j * 4096);
#else
  const uint4* g = (const uint4*)src;
  uint4*       l = (uint4*)dst;
#pragma unroll
  for (int j = 0; j < 4; ++j) l[threadIdx.x + j * 256] = g[threadIdx.x + j * 256];
#endif
}

// copy 512 B (||w||^2 stage) global -> LDS
static __device__ __forceinline__ void stage_wq(float* dst, const float* src) {
#if HAVE_ASYNC_LDS
  if (threadIdx.x < 32)
    async_b128((const char*)src + threadIdx.x * 16, (char*)dst + threadIdx.x * 16);
#else
  if (threadIdx.x < VQ_TC) dst[threadIdx.x] = src[threadIdx.x];
#endif
}

static __device__ __forceinline__ void wait_async_done() {
#if HAVE_ASYNC_LDS
  __builtin_amdgcn_s_wait_asynccnt(0);
#endif
}

// ---------------------------------------------------------------- init
__global__ void k_init(const float* __restrict__ f, float* __restrict__ zrest,
                       float* __restrict__ fhat) {
  int i = blockIdx.x * blockDim.x + threadIdx.x;
  if (i < IMG) { zrest[i] = f[i]; fhat[i] = 0.f; }
}

// ------------------------------------------- codebook -> f16 + ||w||^2
__global__ void k_prep_emb(const float* __restrict__ emb, _Float16* __restrict__ e16,
                           float* __restrict__ wsq) {
  int k = blockIdx.x * blockDim.x + threadIdx.x;
  if (k >= K_) return;
  float s = 0.f;
#pragma unroll
  for (int c = 0; c < C_; ++c) {
    float v = emb[k * C_ + c];
    s += v * v;
    e16[k * C_ + c] = (_Float16)v;
  }
  wsq[k] = s;
}

// ---------------------------- area downsample z_rest -> f16 rows [N][64]
__global__ void k_rowprep(const float* __restrict__ zrest, _Float16* __restrict__ zf16,
                          int pn) {
  int idx = blockIdx.x * blockDim.x + threadIdx.x;
  int N = B_ * pn * pn;
  if (idx >= N * C_) return;
  int c = idx & (C_ - 1);
  int row = idx >> 6;
  int x = row % pn, y = (row / pn) % pn, b = row / (pn * pn);
  int blk = H_ / pn;
  const float* src = zrest + ((b * C_ + c) * H_ + y * blk) * W_ + x * blk;
  float s = 0.f;
  for (int dy = 0; dy < blk; ++dy)
    for (int dx = 0; dx < blk; ++dx) s += src[dy * W_ + dx];
  zf16[row * C_ + c] = (_Float16)(s / (float)(blk * blk));
}

// ------------------------------------- WMMA nearest-codeword search
// 8 waves/block, one 16-row tile per wave. Codebook is staged through a
// double-buffered 16 KB LDS tile (async global->LDS, ASYNCcnt), shared by
// all 8 waves; WMMA consumes it via ds_load. argmin d == argmax(2z.w-||w||^2).
__global__ void __launch_bounds__(256) k_vq(const _Float16* __restrict__ zf16,
                                            const _Float16* __restrict__ e16,
                                            const float* __restrict__ wsq,
                                            int* __restrict__ tok, int nRows) {
  __shared__ alignas(32) _Float16 sb[2][VQ_TC * C_];  // 2 x 16 KB
  __shared__ alignas(16) float    sq[2][VQ_TC];       // 2 x 512 B

  int lane = threadIdx.x & 31;
  int wid  = threadIdx.x >> 5;
  int rt   = blockIdx.x * 8 + wid;
  bool active = (rt * 16 < nRows);    // wave-uniform; barriers stay convergent
  int m    = lane & 15;
  int half = lane >> 4;

  // A tile: 16 rows x K=64 (two 16x32 f16 operands), contiguous k = 16*half + s per lane half.
  // Reads past nRows stay inside the 1 MB zf16 buffer (memory-safe).
  const _Float16* arow = zf16 + (size_t)(rt * 16 + m) * C_;
  v16h a0 = *(const v16h*)(arow +      half * 16);
  v16h a1 = *(const v16h*)(arow + 32 + half * 16);

  float best[8];
  int   bidx[8];
#pragma unroll
  for (int v = 0; v < 8; ++v) { best[v] = -3.0e38f; bidx[v] = 0; }

  stage_cb(&sb[0][0], e16);
  stage_wq(&sq[0][0], wsq);

  for (int s = 0; s < VQ_STAGES; ++s) {
    wait_async_done();
    __syncthreads();                       // stage s resident for whole block
    if (s + 1 < VQ_STAGES) {               // overlap next copy with compute
      stage_cb(&sb[(s + 1) & 1][0], e16 + (size_t)(s + 1) * VQ_TC * C_);
      stage_wq(&sq[(s + 1) & 1][0], wsq + (size_t)(s + 1) * VQ_TC);
    }
    if (active) {
      const _Float16* sbase = &sb[s & 1][0];
#pragma unroll
      for (int ct = 0; ct < VQ_TC / 16; ++ct) {
        int col = s * VQ_TC + ct * 16 + m;
        const _Float16* bp = sbase + (ct * 16 + m) * C_;
        v16h b0 = *(const v16h*)(bp +      half * 16);   // K 0..31
        v16h b1 = *(const v16h*)(bp + 32 + half * 16);   // K 32..63
        v8f c = {};
        c = __builtin_amdgcn_wmma_f32_16x16x32_f16(false, a0, false, b0, (short)0, c, false, false);
        c = __builtin_amdgcn_wmma_f32_16x16x32_f16(false, a1, false, b1, (short)0, c, false, false);
        float wq = sq[s & 1][ct * 16 + m];
#pragma unroll
        for (int v = 0; v < 8; ++v) {
          float sc = 2.f * c[v] - wq;
          if (sc > best[v]) { best[v] = sc; bidx[v] = col; }
        }
      }
    }
    __syncthreads();                       // all reads done before buffer reuse
  }

  if (active) {
#pragma unroll
    for (int v = 0; v < 8; ++v) {
      float s = best[v];
      int   i = bidx[v];
#pragma unroll
      for (int off = 8; off; off >>= 1) {
        float os = __shfl_xor(s, off, 16);
        int   oi = __shfl_xor(i, off, 16);
        if (os > s || (os == s && oi < i)) { s = os; i = oi; }
      }
      if (m == 0) tok[rt * 16 + v + 8 * half] = i;   // lanes 0/16 -> rows v, v+8
    }
  }
}

// ----------------------------- gather codewords + bicubic upsample to 16x16
__device__ __forceinline__ float cubw(float d) {
  d = fabsf(d);
  const float a = -0.5f;                       // Keys kernel (jax cubic)
  if (d <= 1.f) return ((a + 2.f) * d - (a + 3.f)) * d * d + 1.f;
  if (d < 2.f)  return (((d - 5.f) * d + 8.f) * d - 4.f) * a;
  return 0.f;
}

__global__ void k_gather_up(const float* __restrict__ emb, const int* __restrict__ tok,
                            float* __restrict__ hup, int pn) {
  int idx = blockIdx.x * blockDim.x + threadIdx.x;
  if (idx >= IMG) return;
  int x = idx & 15, y = (idx >> 4) & 15, c = (idx >> 8) & 63, b = idx >> 14;
  float val;
  if (pn == 1) {
    val = emb[tok[b] * C_ + c];
  } else {
    float r  = (float)pn / 16.f;
    float sy = ((float)y + 0.5f) * r - 0.5f;
    float sx = ((float)x + 0.5f) * r - 0.5f;
    int iy = (int)floorf(sy), ix = (int)floorf(sx);
    float acc = 0.f;
    for (int j = -1; j <= 2; ++j) {
      float wy = cubw(sy - (float)(iy + j));
      int cy = min(max(iy + j, 0), pn - 1);
      for (int i = -1; i <= 2; ++i) {
        float wx = cubw(sx - (float)(ix + i));
        int cx = min(max(ix + i, 0), pn - 1);
        acc += wy * wx * emb[tok[(b * pn + cy) * pn + cx] * C_ + c];
      }
    }
    val = acc;
  }
  hup[idx] = val;
}

// ----------------------- conv weights -> f16 [co][k], k = (ky*3+kx)*64+ci
__global__ void k_prep_w(const float* __restrict__ phiw, _Float16* __restrict__ wf16,
                         int phi) {
  int idx = blockIdx.x * blockDim.x + threadIdx.x;
  if (idx >= C_ * CONV_K) return;
  int k = idx % CONV_K, co = idx / CONV_K;
  int ci = k & 63, kk = k >> 6, ky = kk / 3, kx = kk % 3;
  float v = phiw[((((size_t)phi * C_ + co) * C_ + ci) * 3 + ky) * 3 + kx];
  wf16[co * CONV_K + k] = (_Float16)v;
}

// ------------------------------------------ im2col (SAME, zero pad) -> f16
__global__ void k_im2col(const float* __restrict__ hup, _Float16* __restrict__ col) {
  int idx = blockIdx.x * blockDim.x + threadIdx.x;
  if (idx >= NPOS * CONV_K) return;
  int k = idx % CONV_K, p = idx / CONV_K;
  int ci = k & 63, kk = k >> 6, ky = kk / 3, kx = kk % 3;
  int x = p & 15, y = (p >> 4) & 15, b = p >> 8;
  int iy = y + ky - 1, ix = x + kx - 1;
  float v = 0.f;
  if (iy >= 0 && iy < 16 && ix >= 0 && ix < 16)
    v = hup[((b * C_ + ci) * 16 + iy) * 16 + ix];
  col[(size_t)p * CONV_K + k] = (_Float16)v;
}

// --------- WMMA conv GEMM [8192 x 576 x 64] + fused phi blend + updates
__global__ void __launch_bounds__(256) k_conv(const _Float16* __restrict__ col,
                                              const _Float16* __restrict__ wf16,
                                              const float* __restrict__ phib,
                                              const float* __restrict__ hup,
                                              float* __restrict__ fhat,
                                              float* __restrict__ zrest, int phi) {
  int lane = threadIdx.x & 31;
  int wid  = threadIdx.x >> 5;
  int rt   = blockIdx.x * 8 + wid;          // 512 row tiles, grid fills exactly
  int m    = lane & 15;
  int half = lane >> 4;
  const _Float16* arow = col + (size_t)(rt * 16 + m) * CONV_K;

  v8f acc[4] = {};
  for (int kc = 0; kc < CONV_K / 32; ++kc) {
    const _Float16* ap = arow + kc * 32;
    if (kc + 1 < CONV_K / 32) __builtin_prefetch(ap + 32, 0, 1);
    v16h a = *(const v16h*)(ap + half * 16);
#pragma unroll
    for (int ct = 0; ct < 4; ++ct) {
      const _Float16* bp = wf16 + (size_t)(ct * 16 + m) * CONV_K + kc * 32 + half * 16;
      v16h b = *(const v16h*)bp;
      acc[ct] = __builtin_amdgcn_wmma_f32_16x16x32_f16(false, a, false, b, (short)0,
                                                       acc[ct], false, false);
    }
  }

#pragma unroll
  for (int ct = 0; ct < 4; ++ct) {
    int co = ct * 16 + m;
    float bias = phib[phi * C_ + co];
#pragma unroll
    for (int v = 0; v < 8; ++v) {
      int row = rt * 16 + v + 8 * half;     // C-matrix row mapping
      int x = row & 15, y = (row >> 4) & 15, b = row >> 8;
      int pos = ((b * C_ + co) * 16 + y) * 16 + x;
      float hv = 0.5f * hup[pos] + 0.5f * (acc[ct][v] + bias);
      fhat[pos]  += hv;
      zrest[pos] -= hv;
    }
  }
}

// ----------------------------------------------------------------- host
extern "C" void kernel_launch(void* const* d_in, const int* in_sizes, int n_in,
                              void* d_out, int out_size, void* d_ws, size_t ws_size,
                              hipStream_t stream) {
  const float* f    = (const float*)d_in[0];
  const float* emb  = (const float*)d_in[1];
  const float* phiw = (const float*)d_in[2];
  const float* phib = (const float*)d_in[3];
  float* fhat = (float*)d_out;
  char* ws = (char*)d_ws;

  float*    zrest = (float*)   (ws + OFF_ZREST);
  _Float16* e16   = (_Float16*)(ws + OFF_EMB16);
  float*    wsq   = (float*)   (ws + OFF_WSQ);
  int*      tok   = (int*)     (ws + OFF_TOK);
  _Float16* wf16  = (_Float16*)(ws + OFF_WF16);
  _Float16* zf16  = (_Float16*)(ws + OFF_ZF16);
  float*    hup   = (float*)   (ws + OFF_HUP);
  _Float16* colb  = (_Float16*)(ws + OFF_COL);

  k_init<<<IMG / 256, 256, 0, stream>>>(f, zrest, fhat);
  k_prep_emb<<<K_ / 256, 256, 0, stream>>>(emb, e16, wsq);

  const int ms[5]  = {1, 2, 4, 8, 16};
  const int phi[5] = {0, 1, 1, 2, 3};   // nearest tick of linspace(1/12, 11/12, 4)
  for (int si = 0; si < 5; ++si) {
    int pn = ms[si];
    int N  = B_ * pn * pn;
    k_rowprep<<<(N * C_ + 255) / 256, 256, 0, stream>>>(zrest, zf16, pn);
    int tiles = N / 16;
    k_vq<<<(tiles + 7) / 8, 256, 0, stream>>>(zf16, e16, wsq, tok, N);
    k_gather_up<<<IMG / 256, 256, 0, stream>>>(emb, tok, hup, pn);
    k_prep_w<<<(C_ * CONV_K + 255) / 256, 256, 0, stream>>>(phiw, wf16, phi[si]);
    k_im2col<<<(NPOS * CONV_K + 255) / 256, 256, 0, stream>>>(hup, colb);
    k_conv<<<NPOS / 16 / 8, 256, 0, stream>>>(colb, wf16, phib, hup, fhat, zrest, phi[si]);
  }
  (void)in_sizes; (void)n_in; (void)out_size; (void)ws_size;
}
